// ReActConv2d_67027259621656
// MI455X (gfx1250) — 8-bit matrix operand order repaired and checked by reading, compiled, not run on hardware
//
#include <hip/hip_runtime.h>
#include <stdint.h>

#define N_  32
#define C_  256
#define H_  56
#define W_  56
#define O_  256
#define HP  58
#define WP  58
#define HW_ (H_ * W_)
#define PIX (N_ * HW_)

typedef __attribute__((ext_vector_type(8))) int v8i;
typedef __attribute__((ext_vector_type(4))) int v4i;
typedef __attribute__((ext_vector_type(2))) int v2i;

typedef __attribute__((address_space(1))) v4i v4i_as1;
typedef __attribute__((address_space(3))) v4i v4i_as3;

#if defined(__has_builtin)
#if __has_builtin(__builtin_amdgcn_global_load_async_to_lds_b128)
#define HAVE_ASYNC_B128 1
#endif
#if __has_builtin(__builtin_amdgcn_s_wait_asynccnt)
#define HAVE_WAIT_ASYNC 1
#endif
#endif

__device__ __forceinline__ void stage16(const int8_t* g, int8_t* l) {
#if defined(HAVE_ASYNC_B128)
  __builtin_amdgcn_global_load_async_to_lds_b128(
      (v4i_as1*)(void*)g, (v4i_as3*)(void*)l, 0, 0);
#else
  *(int4*)l = *(const int4*)g;
#endif
}

template <int CNT>
__device__ __forceinline__ void async_wait_n() {
#if defined(HAVE_ASYNC_B128)
#if defined(HAVE_WAIT_ASYNC)
  __builtin_amdgcn_s_wait_asynccnt(CNT);
#else
  asm volatile("s_wait_asynccnt %0" ::"i"(CNT) : "memory");
#endif
#endif
}

__global__ __launch_bounds__(256) void binarize_act_kernel(
    const float* __restrict__ x, const float* __restrict__ bias0,
    int8_t* __restrict__ bapad) {
  const int total = N_ * HP * WP * C_;
  for (int i = blockIdx.x * blockDim.x + threadIdx.x; i < total;
       i += gridDim.x * blockDim.x) {
    const int c = i & (C_ - 1);
    int t = i >> 8;
    const int wp = t % WP;
    t /= WP;
    const int hp = t % HP;
    const int n = t / HP;
    int8_t v = 0;
    if (hp >= 1 && hp <= H_ && wp >= 1 && wp <= W_) {
      float s = x[((n * C_ + c) * H_ + (hp - 1)) * W_ + (wp - 1)] + bias0[c];
      v = (int8_t)((s > 0.f) - (s < 0.f));
    }
    bapad[i] = v;
  }
}

__global__ __launch_bounds__(256) void binarize_wgt_kernel(
    const float* __restrict__ wgt, int8_t* __restrict__ bwt,
    float* __restrict__ scale) {
  __shared__ float red[256];
  const int o = blockIdx.x;
  const int c = threadIdx.x;
  float s = 0.f;
#pragma unroll
  for (int r = 0; r < 9; ++r) {
    float w = wgt[(o * C_ + c) * 9 + r];
    s += fabsf(w);
    bwt[(r * O_ + o) * C_ + c] = (int8_t)((w > 0.f) - (w < 0.f));
  }
  red[c] = s;
  __syncthreads();
  for (int off = 128; off > 0; off >>= 1) {
    if (c < off) red[c] += red[c + off];
    __syncthreads();
  }
  if (c == 0) scale[o] = red[0] / 2304.f;
}

__global__ __launch_bounds__(256) void breact_gemm_kernel(
    const int8_t* __restrict__ bapad, const int8_t* __restrict__ bwt,
    const float* __restrict__ scale, float* __restrict__ out) {
  __shared__ __attribute__((aligned(32))) int8_t lA[2][8192];
  __shared__ __attribute__((aligned(32))) int8_t lB[2][8192];
  __shared__ float lS[128];

  const int tid = threadIdx.x;
  const int lane = tid & 31;
  const int laneLo = lane & 15;
  const int laneHi = lane >> 4;
  const int wid = tid >> 5;
  const int waveM = wid & 1;
  const int waveN = wid >> 1;

  const int mtile = blockIdx.x & 1;
  const int ptile = blockIdx.x >> 1;
  const int mbase = mtile * 128;
  const int pbase = ptile * 128;

  if (tid < 128) lS[tid] = scale[mbase + tid];

  const int rowS = tid & 127;
  const int half = tid >> 7;
  const int pixB = pbase + rowS;
  const int nB = pixB / HW_;
  const int hwB = pixB - nB * HW_;
  const int hB = hwB / W_;
  const int wB = hwB - hB * W_;
  const int pixBase = ((nB * HP + hB) * WP + wB) * C_;

  v8i acc[4][2];
#pragma unroll
  for (int i = 0; i < 4; ++i)
#pragma unroll
    for (int j = 0; j < 2; ++j) acc[i][j] = (v8i){0, 0, 0, 0, 0, 0, 0, 0};

  auto stageStep = [&](int step, int buf) {
    const int r = step >> 2;
    const int cb = step & 3;
    const int kh = r / 3, kw = r % 3;
    const int koff = cb * 64 + half * 32;
    const int ldst = rowS * 64 + half * 32;
    const int8_t* sA = bwt + (r * O_ + mbase + rowS) * C_ + koff;
    stage16(sA, &lA[buf][ldst]);
    stage16(sA + 16, &lA[buf][ldst + 16]);
    const int8_t* sB = bapad + pixBase + (kh * WP + kw) * C_ + koff;
    stage16(sB, &lB[buf][ldst]);
    stage16(sB + 16, &lB[buf][ldst + 16]);
  };

  auto computeStep = [&](int buf) {
    v8i af[4], bf[2];
#pragma unroll
    for (int mf = 0; mf < 4; ++mf) {
      const int base = (waveM * 64 + mf * 16 + laneLo) * 64 + laneHi * 8;
      v2i c0 = *(const v2i*)&lA[buf][base + 0];
      v2i c1 = *(const v2i*)&lA[buf][base + 16];
      v2i c2 = *(const v2i*)&lA[buf][base + 32];
      v2i c3 = *(const v2i*)&lA[buf][base + 48];
      af[mf] = (v8i){c0[0], c0[1], c1[0], c1[1], c2[0], c2[1], c3[0], c3[1]};
    }
#pragma unroll
    for (int nf = 0; nf < 2; ++nf) {
      const int base = (waveN * 32 + nf * 16 + laneLo) * 64 + laneHi * 8;
      v2i d0 = *(const v2i*)&lB[buf][base + 0];
      v2i d1 = *(const v2i*)&lB[buf][base + 16];
      v2i d2 = *(const v2i*)&lB[buf][base + 32];
      v2i d3 = *(const v2i*)&lB[buf][base + 48];
      bf[nf] = (v8i){d0[0], d0[1], d1[0], d1[1], d2[0], d2[1], d3[0], d3[1]};
    }
#pragma unroll
    for (int mf = 0; mf < 4; ++mf)
#pragma unroll
      for (int nf = 0; nf < 2; ++nf)
        acc[mf][nf] = __builtin_amdgcn_wmma_i32_16x16x64_iu8(
            true, af[mf], true, bf[nf], acc[mf][nf], false, false);
  };

  stageStep(0, 0);
#pragma unroll 2
  for (int s = 0; s < 36; ++s) {
    const int buf = s & 1;
    if (s + 1 < 36) {
      stageStep(s + 1, buf ^ 1);
      async_wait_n<4>();
    } else {
      async_wait_n<0>();
    }
    __syncthreads();
    computeStep(buf);
    __syncthreads();
  }

#pragma unroll
  for (int nf = 0; nf < 2; ++nf) {
    const int p = pbase + waveN * 32 + nf * 16 + laneLo;
    const int n = p / HW_;
    const int hw = p - n * HW_;
    const int h = hw / W_;
    const int w = hw - h * W_;
#pragma unroll
    for (int mf = 0; mf < 4; ++mf) {
#pragma unroll
      for (int j = 0; j < 8; ++j) {
        const int ml = waveM * 64 + mf * 16 + j + (laneHi << 3);
        const float v = (float)acc[mf][nf][j] * lS[ml];
        out[((n * O_ + (mbase + ml)) * H_ + h) * W_ + w] = v;
      }
    }
  }
}

extern "C" void kernel_launch(void* const* d_in, const int* in_sizes, int n_in,
                              void* d_out, int out_size, void* d_ws,
                              size_t ws_size, hipStream_t stream) {
  const float* x = (const float*)d_in[0];
  const float* wgt = (const float*)d_in[1];
  const float* bias0 = (const float*)d_in[2];
  float* out = (float*)d_out;

  uint8_t* ws = (uint8_t*)d_ws;
  size_t off = 0;
  int8_t* bapad = (int8_t*)(ws + off);
  off += (size_t)N_ * HP * WP * C_;
  off = (off + 255) & ~(size_t)255;
  int8_t* bwt = (int8_t*)(ws + off);
  off += (size_t)9 * O_ * C_;
  off = (off + 255) & ~(size_t)255;
  float* scale = (float*)(ws + off);

  binarize_act_kernel<<<8192, 256, 0, stream>>>(x, bias0, bapad);
  binarize_wgt_kernel<<<O_, 256, 0, stream>>>(wgt, bwt, scale);
  breact_gemm_kernel<<<(O_ / 128) * (PIX / 128), 256, 0, stream>>>(bapad, bwt,
                                                                   scale, out);
}
